// Attention_6201932775664
// MI455X (gfx1250) — matrix operand order repaired and checked by reading, not compiled, not run on hardware
//
#include <hip/hip_runtime.h>
#include <hip/hip_bf16.h>

typedef _Float16 f16;
typedef __attribute__((ext_vector_type(16))) _Float16 v16h;
typedef __attribute__((ext_vector_type(8)))  _Float16 v8h;
typedef __attribute__((ext_vector_type(8)))  float    v8f;
typedef __attribute__((ext_vector_type(2)))  float    f32x2;

#define DIM    1024
#define HEADS  16
#define DHEAD  64
#define INNER  1024
#define NCTX   2048
#define BATCH  2
#define MROWS  (BATCH * NCTX)   // 4096

// ---------------------------------------------------------------------------
// WMMA fragment helpers (wave32, A/B k order is this code's contiguous choice)
// A (16x32 f16): lane L holds row r=L&15; halves p = K (16*hi)+p
//                (hi = L>>4)
// B (32x16 f16): lane L holds col c=L&15; halves p = K (16*hi)+p; operand is
//                stored transposed (N x K row-major) -> one 32B chunk per lane.
// C/D (16x16 f32): lane holds col c=L&15, VGPR v holds row v + 8*hi.
// ---------------------------------------------------------------------------

__device__ __forceinline__ v16h load_a_frag(const f16* base, int ld, int kBase) {
  int lane = threadIdx.x & 31;
  int r = lane & 15, hi = lane >> 4;
  const f16* p = base + (size_t)r * ld + kBase + 16 * hi;
  v8h lo = *(const v8h*)(p);
  v8h hv = *(const v8h*)(p + 8);
  v16h a;
#pragma unroll
  for (int t = 0; t < 8; ++t) { a[t] = lo[t]; a[8 + t] = hv[t]; }
  return a;
}

__device__ __forceinline__ v16h load_b_frag(const f16* baseT, int ld) {
  int lane = threadIdx.x & 31;
  int c = lane & 15, hi = lane >> 4;
  return *(const v16h*)(baseT + (size_t)c * ld + 16 * hi);
}

__device__ __forceinline__ v8f wmma_f16(v16h a, v16h b, v8f c) {
  return __builtin_amdgcn_wmma_f32_16x16x32_f16(false, a, false, b,
                                                (short)0, c, false, false);
}

// ---------------------------------------------------------------------------
// LayerNorm: fp32 in -> f16 out, one 256-thread block per row of 1024
// ---------------------------------------------------------------------------
__global__ void ln_kernel(const float* __restrict__ x, const float* __restrict__ g,
                          const float* __restrict__ bt, f16* __restrict__ xn) {
  int row = blockIdx.x;
  int t = threadIdx.x;
  float4 xv = ((const float4*)(x + (size_t)row * DIM))[t];
  float s  = xv.x + xv.y + xv.z + xv.w;
  float s2 = xv.x * xv.x + xv.y * xv.y + xv.z * xv.z + xv.w * xv.w;
  __shared__ float r1[256], r2[256];
  r1[t] = s; r2[t] = s2; __syncthreads();
  for (int o = 128; o > 0; o >>= 1) {
    if (t < o) { r1[t] += r1[t + o]; r2[t] += r2[t + o]; }
    __syncthreads();
  }
  float mu  = r1[0] * (1.0f / DIM);
  float var = r2[0] * (1.0f / DIM) - mu * mu;
  float rstd = rsqrtf(var + 1e-5f);
  float4 gv = ((const float4*)g)[t];
  float4 bv = ((const float4*)bt)[t];
  f16* o = xn + (size_t)row * DIM + t * 4;
  o[0] = (f16)((xv.x - mu) * rstd * gv.x + bv.x);
  o[1] = (f16)((xv.y - mu) * rstd * gv.y + bv.y);
  o[2] = (f16)((xv.z - mu) * rstd * gv.z + bv.z);
  o[3] = (f16)((xv.w - mu) * rstd * gv.w + bv.w);
}

// ---------------------------------------------------------------------------
// Weight repacking: fused [Wq | Wk | Wv] transposed to [3072][1024] f16
// ---------------------------------------------------------------------------
__global__ void pack_qkvT(const float* __restrict__ Wq, const float* __restrict__ Wkv,
                          f16* __restrict__ wT) {
  int idx = blockIdx.x * 256 + threadIdx.x;  // idx = c*1024 + k
  int c = idx >> 10;
  int k = idx & 1023;
  float v;
  if (c < 1024)      v = Wq [(size_t)k * 1024 + c];
  else if (c < 2048) v = Wkv[(size_t)k * 2048 + (c - 1024)];
  else               v = Wkv[(size_t)k * 2048 + (c - 2048) + 1024];
  wT[idx] = (f16)v;
}

__global__ void pack_wT(const float* __restrict__ W, f16* __restrict__ wT) {
  int idx = blockIdx.x * 256 + threadIdx.x;  // idx = c*1024 + k
  int c = idx >> 10;
  int k = idx & 1023;
  wT[idx] = (f16)W[(size_t)k * 1024 + c];
}

// ---------------------------------------------------------------------------
// QKV projection GEMM: 32x64 tile per wave, double-buffered A and B frags.
// Scatter to q/k/vT head-major layouts; softmax scale folded into q.
// ---------------------------------------------------------------------------
__global__ void gemm_qkv(const f16* __restrict__ xn, const f16* __restrict__ wT,
                         f16* __restrict__ q, f16* __restrict__ k,
                         f16* __restrict__ vT) {
  int rowBase = blockIdx.x * 32;
  int colBase = blockIdx.y * 64;
  v8f acc[2][4] = {};
  const f16* a0 = xn + (size_t)rowBase * DIM;
  const f16* a1 = a0 + (size_t)16 * DIM;
  v16h A0 = load_a_frag(a0, DIM, 0);
  v16h A1 = load_a_frag(a1, DIM, 0);
  v16h B[4];
#pragma unroll
  for (int s = 0; s < 4; ++s)
    B[s] = load_b_frag(wT + (size_t)(colBase + s * 16) * DIM, DIM);

  for (int k0 = 0; k0 < DIM; k0 += 32) {
    int kn = (k0 + 32 < DIM) ? k0 + 32 : 0;   // wrap: harmless redundant load
    v16h Bn[4];
#pragma unroll
    for (int s = 0; s < 4; ++s)
      Bn[s] = load_b_frag(wT + (size_t)(colBase + s * 16) * DIM + kn, DIM);
    v16h An0 = load_a_frag(a0, DIM, kn);
    v16h An1 = load_a_frag(a1, DIM, kn);
#pragma unroll
    for (int s = 0; s < 4; ++s) {
      acc[0][s] = wmma_f16(A0, B[s], acc[0][s]);
      acc[1][s] = wmma_f16(A1, B[s], acc[1][s]);
    }
    A0 = An0; A1 = An1;
#pragma unroll
    for (int s = 0; s < 4; ++s) B[s] = Bn[s];
  }

  int lane = threadIdx.x & 31;
  int col = lane & 15, hi = lane >> 4;
#pragma unroll
  for (int rt = 0; rt < 2; ++rt) {
#pragma unroll
    for (int s = 0; s < 4; ++s) {
      int gc = colBase + s * 16 + col;
      int sec = gc >> 10;
      int within = gc & 1023;
      int head = within >> 6;
      int dd = within & 63;
#pragma unroll
      for (int v = 0; v < 8; ++v) {
        int m = rowBase + rt * 16 + v + 8 * hi;
        int bb = m >> 11, i = m & 2047;
        float val = acc[rt][s][v];
        size_t hb = (size_t)(bb * HEADS + head);
        if (sec == 0)      q [(hb * NCTX + i) * DHEAD + dd] = (f16)(val * 0.125f);
        else if (sec == 1) k [(hb * NCTX + i) * DHEAD + dd] = (f16)val;
        else               vT[(hb * DHEAD + dd) * NCTX + i] = (f16)val;
      }
    }
  }
}

// ---------------------------------------------------------------------------
// Fused talking-heads attention. One workgroup = 16 waves = 16 heads for one
// 16-row query tile. Two-pass softmax; head mixing via LDS in float2; K frags
// double-buffered across j-tiles; V frags loaded early each tile.
// ---------------------------------------------------------------------------
__global__ __launch_bounds__(512) void attn_kernel(
    const f16* __restrict__ q, const f16* __restrict__ k,
    const f16* __restrict__ vT, const float* __restrict__ mixp,
    const float* __restrict__ mixq, f16* __restrict__ ao) {
  __shared__ float smp[256], smq[256];
  __shared__ float lds_dots[HEADS][16][16][2];   // [h][row][col][js]
  __shared__ float lds_attn[HEADS][16][16][2];
  __shared__ f16   lds_am[HEADS][16 * 32];

  int tid = threadIdx.x;
  int w = tid >> 5;                 // wave id: head h / out-head g / g2 roles
  int lane = tid & 31;
  int col = lane & 15, hi = lane >> 4;
  int iBase = blockIdx.x * 16;
  int bb = blockIdx.y;

  if (tid < 256) smp[tid] = mixp[tid];
  else           smq[tid - 256] = mixq[tid - 256];
  __syncthreads();

  const f16* qbase = q + ((size_t)(bb * HEADS + w) * NCTX + iBase) * DHEAD;
  v16h qa0 = load_a_frag(qbase, DHEAD, 0);
  v16h qa1 = load_a_frag(qbase, DHEAD, 32);
  const f16* kbase = k + (size_t)(bb * HEADS + w) * NCTX * DHEAD;
  const f16* vbase = vT + (size_t)(bb * HEADS + w) * DHEAD * NCTX;

  float mrow[8], lrow[8];
#pragma unroll
  for (int v = 0; v < 8; ++v) { mrow[v] = -1e30f; lrow[v] = 0.f; }

  // K fragments for j-tile 0 (double buffer lives in kb[])
  v16h kb[4];   // [js*2 + kd]
#pragma unroll
  for (int js = 0; js < 2; ++js) {
    kb[js * 2 + 0] = load_b_frag(kbase + (size_t)(js * 16) * DHEAD, DHEAD);
    kb[js * 2 + 1] = load_b_frag(kbase + (size_t)(js * 16) * DHEAD + 32, DHEAD);
  }

  // ---------------- Pass 1: streaming softmax statistics ----------------
  for (int j0 = 0; j0 < NCTX; j0 += 32) {
#pragma unroll
    for (int js = 0; js < 2; ++js) {
      v8f c = {};
      c = wmma_f16(qa0, kb[js * 2 + 0], c);
      c = wmma_f16(qa1, kb[js * 2 + 1], c);
#pragma unroll
      for (int v = 0; v < 8; ++v) lds_dots[w][v + 8 * hi][col][js] = c[v];
    }
    {  // prefetch next tile's K frags (wraps to 0 -> reused by pass 2)
      int j1 = (j0 + 32 < NCTX) ? j0 + 32 : 0;
#pragma unroll
      for (int js = 0; js < 2; ++js) {
        kb[js * 2 + 0] = load_b_frag(kbase + (size_t)(j1 + js * 16) * DHEAD, DHEAD);
        kb[js * 2 + 1] = load_b_frag(kbase + (size_t)(j1 + js * 16) * DHEAD + 32, DHEAD);
      }
    }
    __syncthreads();
#pragma unroll
    for (int v = 0; v < 8; ++v) {
      int r = v + 8 * hi;
      f32x2 s = {0.f, 0.f};
#pragma unroll
      for (int h = 0; h < HEADS; ++h)
        s += smp[h * HEADS + w] * (*(const f32x2*)&lds_dots[h][r][col][0]);
      float t = fmaxf(s[0], s[1]);
#pragma unroll
      for (int msk = 8; msk > 0; msk >>= 1) t = fmaxf(t, __shfl_xor(t, msk, 32));
      float mnew = fmaxf(mrow[v], t);
      float p = __expf(s[0] - mnew) + __expf(s[1] - mnew);
#pragma unroll
      for (int msk = 8; msk > 0; msk >>= 1) p += __shfl_xor(p, msk, 32);
      lrow[v] = lrow[v] * __expf(mrow[v] - mnew) + p;
      mrow[v] = mnew;
    }
    __syncthreads();
  }
  float invl[8];
#pragma unroll
  for (int v = 0; v < 8; ++v) invl[v] = 1.0f / lrow[v];

  // ---------------- Pass 2: normalized probs -> post-mix -> PV ----------------
  v8f oacc[4] = {};
  for (int j0 = 0; j0 < NCTX; j0 += 32) {
    // V fragments for this tile, issued early; consumed after the barriers
    v16h vb[4];
#pragma unroll
    for (int dt = 0; dt < 4; ++dt)
      vb[dt] = load_b_frag(vbase + (size_t)(dt * 16) * NCTX + j0, NCTX);
#pragma unroll
    for (int js = 0; js < 2; ++js) {
      v8f c = {};
      c = wmma_f16(qa0, kb[js * 2 + 0], c);
      c = wmma_f16(qa1, kb[js * 2 + 1], c);
#pragma unroll
      for (int v = 0; v < 8; ++v) lds_dots[w][v + 8 * hi][col][js] = c[v];
    }
    {  // prefetch next tile's K frags
      int j1 = (j0 + 32 < NCTX) ? j0 + 32 : 0;
#pragma unroll
      for (int js = 0; js < 2; ++js) {
        kb[js * 2 + 0] = load_b_frag(kbase + (size_t)(j1 + js * 16) * DHEAD, DHEAD);
        kb[js * 2 + 1] = load_b_frag(kbase + (size_t)(j1 + js * 16) * DHEAD + 32, DHEAD);
      }
    }
    __syncthreads();
#pragma unroll
    for (int v = 0; v < 8; ++v) {
      int r = v + 8 * hi;
      f32x2 s = {0.f, 0.f};
#pragma unroll
      for (int h = 0; h < HEADS; ++h)
        s += smp[h * HEADS + w] * (*(const f32x2*)&lds_dots[h][r][col][0]);
      lds_attn[w][r][col][0] = __expf(s[0] - mrow[v]) * invl[v];
      lds_attn[w][r][col][1] = __expf(s[1] - mrow[v]) * invl[v];
    }
    __syncthreads();
#pragma unroll
    for (int v = 0; v < 8; ++v) {
      int r = v + 8 * hi;
      f32x2 a = {0.f, 0.f};
#pragma unroll
      for (int g = 0; g < HEADS; ++g)
        a += smq[g * HEADS + w] * (*(const f32x2*)&lds_attn[g][r][col][0]);
      lds_am[w][r * 32 + col]      = (f16)a[0];
      lds_am[w][r * 32 + col + 16] = (f16)a[1];
    }
    __syncthreads();
    v16h A = load_a_frag(&lds_am[w][0], 32, 0);
#pragma unroll
    for (int dt = 0; dt < 4; ++dt) oacc[dt] = wmma_f16(A, vb[dt], oacc[dt]);
  }

#pragma unroll
  for (int dt = 0; dt < 4; ++dt) {
#pragma unroll
    for (int v = 0; v < 8; ++v) {
      int r = v + 8 * hi;
      size_t m = (size_t)bb * NCTX + iBase + r;
      ao[m * INNER + w * DHEAD + dt * 16 + col] = (f16)oacc[dt][v];
    }
  }
}

// ---------------------------------------------------------------------------
// Output GEMM: 32x64 tile per wave, double-buffered; ao @ Wout + bout -> fp32
// ---------------------------------------------------------------------------
__global__ void gemm_out(const f16* __restrict__ ao, const f16* __restrict__ wT,
                         const float* __restrict__ bout, float* __restrict__ out) {
  int rowBase = blockIdx.x * 32;
  int colBase = blockIdx.y * 64;
  v8f acc[2][4] = {};
  const f16* a0 = ao + (size_t)rowBase * DIM;
  const f16* a1 = a0 + (size_t)16 * DIM;
  v16h A0 = load_a_frag(a0, DIM, 0);
  v16h A1 = load_a_frag(a1, DIM, 0);
  v16h B[4];
#pragma unroll
  for (int s = 0; s < 4; ++s)
    B[s] = load_b_frag(wT + (size_t)(colBase + s * 16) * DIM, DIM);

  for (int k0 = 0; k0 < DIM; k0 += 32) {
    int kn = (k0 + 32 < DIM) ? k0 + 32 : 0;
    v16h Bn[4];
#pragma unroll
    for (int s = 0; s < 4; ++s)
      Bn[s] = load_b_frag(wT + (size_t)(colBase + s * 16) * DIM + kn, DIM);
    v16h An0 = load_a_frag(a0, DIM, kn);
    v16h An1 = load_a_frag(a1, DIM, kn);
#pragma unroll
    for (int s = 0; s < 4; ++s) {
      acc[0][s] = wmma_f16(A0, B[s], acc[0][s]);
      acc[1][s] = wmma_f16(A1, B[s], acc[1][s]);
    }
    A0 = An0; A1 = An1;
#pragma unroll
    for (int s = 0; s < 4; ++s) B[s] = Bn[s];
  }

  int lane = threadIdx.x & 31;
  int col = lane & 15, hi = lane >> 4;
#pragma unroll
  for (int rt = 0; rt < 2; ++rt) {
#pragma unroll
    for (int s = 0; s < 4; ++s) {
      int gc = colBase + s * 16 + col;
      float bv = bout[gc];
#pragma unroll
      for (int v = 0; v < 8; ++v) {
        int m = rowBase + rt * 16 + v + 8 * hi;
        out[(size_t)m * DIM + gc] = acc[rt][s][v] + bv;
      }
    }
  }
}

// ---------------------------------------------------------------------------
extern "C" void kernel_launch(void* const* d_in, const int* in_sizes, int n_in,
                              void* d_out, int out_size, void* d_ws, size_t ws_size,
                              hipStream_t stream) {
  (void)in_sizes; (void)n_in; (void)out_size; (void)ws_size;
  const float* x    = (const float*)d_in[0];
  const float* ln_g = (const float*)d_in[1];
  const float* ln_b = (const float*)d_in[2];
  const float* Wq   = (const float*)d_in[3];
  const float* Wkv  = (const float*)d_in[4];
  const float* mixp = (const float*)d_in[5];
  const float* mixq = (const float*)d_in[6];
  const float* Wout = (const float*)d_in[7];
  const float* bout = (const float*)d_in[8];
  float* out = (float*)d_out;

  f16* p = (f16*)d_ws;
  f16* xn16  = p; p += (size_t)MROWS * DIM;                 // 8 MB
  f16* qkvT  = p; p += (size_t)3 * INNER * DIM;             // 6 MB
  f16* woutT = p; p += (size_t)INNER * DIM;                 // 2 MB
  f16* q16   = p; p += (size_t)BATCH * HEADS * NCTX * DHEAD;// 8 MB
  f16* k16   = p; p += (size_t)BATCH * HEADS * NCTX * DHEAD;// 8 MB
  f16* vT16  = p; p += (size_t)BATCH * HEADS * NCTX * DHEAD;// 8 MB
  f16* ao16  = p; p += (size_t)MROWS * INNER;               // 8 MB

  ln_kernel<<<MROWS, 256, 0, stream>>>(x, ln_g, ln_b, xn16);
  pack_qkvT<<<(3 * INNER * DIM) / 256, 256, 0, stream>>>(Wq, Wkv, qkvT);
  pack_wT<<<(INNER * DIM) / 256, 256, 0, stream>>>(Wout, woutT);
  gemm_qkv<<<dim3(MROWS / 32, (3 * INNER) / 64), 32, 0, stream>>>(xn16, qkvT, q16, k16, vT16);
  attn_kernel<<<dim3(NCTX / 16, BATCH), 512, 0, stream>>>(q16, k16, vT16, mixp, mixq, ao16);
  gemm_out<<<dim3(MROWS / 32, DIM / 64), 32, 0, stream>>>(ao16, woutT, bout, out);
}
